// MultiHeadAttention_88115549045436
// MI455X (gfx1250) — matrix operand order repaired and checked by reading, not compiled, not run on hardware
//
#include <hip/hip_runtime.h>
#include <cstdint>

// ---------------------------------------------------------------------------
// MHA forward for MI455X (gfx1250, wave32, WMMA).
// B=2, S=2048, D=1024, H=16, HD=64. Softmax over the QUERY axis (axis=-2).
// bf16 WMMA (16x16x32, f32 accum) for all GEMMs; two-pass column softmax to
// avoid the 536 MB score tensor; async Global->LDS copies for bf16 tiles.
// ---------------------------------------------------------------------------

#define DEV static __device__ __forceinline__

typedef __attribute__((ext_vector_type(16))) __bf16 v16bf;
typedef __attribute__((ext_vector_type(8)))  float  v8f;

constexpr int Bc  = 2;
constexpr int Sc  = 2048;
constexpr int Dc  = 1024;
constexpr int Hc  = 16;
constexpr int HDc = 64;

// ---- scalar helpers -------------------------------------------------------

DEV unsigned short f2bfu(float f) {          // fp32 -> bf16 bits, RNE
    unsigned int u = __builtin_bit_cast(unsigned int, f);
    u = u + 0x7FFFu + ((u >> 16) & 1u);
    return (unsigned short)(u >> 16);
}
DEV __bf16 u16bf(unsigned short u) { return __builtin_bit_cast(__bf16, u); }

DEV v8f vzero8() {
    v8f z;
#pragma unroll
    for (int i = 0; i < 8; ++i) z[i] = 0.0f;
    return z;
}

DEV v8f wmma_bf16(v16bf a, v16bf b, v8f c) {
    // D = A(16x32) x B(32x16) + C, f32 accumulate.
    return __builtin_amdgcn_wmma_f32_16x16x32_bf16(
        /*neg_a=*/false, a, /*neg_b=*/false, b,
        /*c_mod=*/(short)0, c, /*reuse_a=*/false, /*reuse_b=*/false);
}

// ---- CDNA5 async Global->LDS (ASYNCcnt-tracked) ---------------------------
// Per-lane 16B transfer: LDS[lds_off + ...] = MEM[gptr].  lds_off is the
// 32-bit offset within the workgroup LDS allocation (low 32 bits of the
// generic pointer to a __shared__ object).
DEV void async_ld_b128(void* lds_ptr, const void* gptr) {
    unsigned int lds_off = (unsigned int)(size_t)lds_ptr;
    asm volatile("global_load_async_to_lds_b128 %0, %1, off"
                 :
                 : "v"(lds_off), "v"(gptr)
                 : "memory");
}
DEV void wait_async0() {
    asm volatile("s_wait_asynccnt 0" ::: "memory");
}

// ---- WMMA fragment gathers from LDS (bf16 bits stored as u16) -------------
// A fragment, 16x32 (MxK). Layout used here (contiguous k, same as B loaders): lane L holds row M=L&15; lane half
// H=L>>4 selects K-subgroup. half h -> K = 16*H + h.
DEV v16bf load_frag_a(const unsigned short* lds, int m0, int ld, int k0, int lane) {
    const int M = m0 + (lane & 15);
    const int Hh = lane >> 4;
    const unsigned short* p = lds + M * ld + k0;
    v16bf a;
#pragma unroll
    for (int h = 0; h < 16; ++h) {
        int K = Hh * 16 + h;
        a[h] = u16bf(p[K]);
    }
    return a;
}

// B fragment, 32x16 (KxN), source tile stored [n][k] (k contiguous per n).
// Layout used here (same contiguous k order as A): lane L -> N=L&15, K = (L>>4)*16 + h.
DEV v16bf load_frag_b_nk(const unsigned short* lds, int n0, int ld, int k0, int lane) {
    const int N = n0 + (lane & 15);
    const unsigned short* p = lds + N * ld + k0 + (lane >> 4) * 16;
    v16bf b;
#pragma unroll
    for (int h = 0; h < 16; ++h) b[h] = u16bf(p[h]);
    return b;
}

// B fragment, 32x16 (KxN), source tile stored [k][n] (n contiguous per k).
DEV v16bf load_frag_b_kn(const unsigned short* lds, int k0, int ld, int n0, int lane) {
    const int N = n0 + (lane & 15);
    const int kk = k0 + (lane >> 4) * 16;
    v16bf b;
#pragma unroll
    for (int h = 0; h < 16; ++h) b[h] = u16bf(lds[(kk + h) * ld + N]);
    return b;
}

// ===========================================================================
// Kernel 1: fused QKV projection.  out = x @ W^T + b, written bf16 into
// head-major [B,H,S,HD].  Block tile 128(M) x 64(N), K-step 32.
// grid = (N/64=16, M/128=32, 3), block = 256 (8 waves, 4x2 wave grid).
// ===========================================================================
__global__ __launch_bounds__(256) void k_qkv(
    const float* __restrict__ x,
    const float* __restrict__ wq, const float* __restrict__ bq,
    const float* __restrict__ wk, const float* __restrict__ bk,
    const float* __restrict__ wv, const float* __restrict__ bv,
    unsigned short* __restrict__ qo, unsigned short* __restrict__ ko,
    unsigned short* __restrict__ vo)
{
    __shared__ unsigned short As[128 * 32];
    __shared__ unsigned short Bs[64 * 32];

    const int tid  = threadIdx.x;
    const int lane = tid & 31;
    const int wave = tid >> 5;
    const int wm   = wave >> 1;          // 0..3  -> 32 rows each
    const int wn   = wave & 1;           // 0..1  -> 32 cols each
    const int n0   = blockIdx.x * 64;
    const int m0   = blockIdx.y * 128;
    const int mat  = blockIdx.z;

    const float* w    = (mat == 0) ? wq : (mat == 1) ? wk : wv;
    const float* bias = (mat == 0) ? bq : (mat == 1) ? bk : bv;
    unsigned short* outp = (mat == 0) ? qo : (mat == 1) ? ko : vo;

    v8f acc[2][2];
#pragma unroll
    for (int mt = 0; mt < 2; ++mt)
#pragma unroll
        for (int nt = 0; nt < 2; ++nt) acc[mt][nt] = vzero8();

    const int arow = tid >> 1, ahalf = tid & 1;   // A stage: 2 threads/row
    const int brow = tid >> 2, bq4   = tid & 3;   // B stage: 4 threads/row

    for (int k0 = 0; k0 < Dc; k0 += 32) {
        // warm L2/WGP$ for next K-step (global_prefetch_b8)
        if (k0 + 32 < Dc) {
            __builtin_prefetch(&x[(size_t)(m0 + arow) * Dc + k0 + 32 + ahalf * 16], 0, 3);
            __builtin_prefetch(&w[(size_t)(n0 + brow) * Dc + k0 + 32 + bq4 * 8], 0, 3);
        }
        // stage A = x[m0..m0+128, k0..k0+32) as bf16, [row][k]
#pragma unroll
        for (int i = 0; i < 4; ++i) {
            int c = ahalf * 16 + i * 4;
            float4 t = *reinterpret_cast<const float4*>(&x[(size_t)(m0 + arow) * Dc + k0 + c]);
            unsigned int lo = f2bfu(t.x) | ((unsigned int)f2bfu(t.y) << 16);
            unsigned int hi = f2bfu(t.z) | ((unsigned int)f2bfu(t.w) << 16);
            *reinterpret_cast<uint2*>(&As[arow * 32 + c]) = make_uint2(lo, hi);
        }
        // stage Bt[n][k] = W[n0+n][k0+k]  (row of W is contiguous in k)
#pragma unroll
        for (int i = 0; i < 2; ++i) {
            int c = bq4 * 8 + i * 4;
            float4 t = *reinterpret_cast<const float4*>(&w[(size_t)(n0 + brow) * Dc + k0 + c]);
            unsigned int lo = f2bfu(t.x) | ((unsigned int)f2bfu(t.y) << 16);
            unsigned int hi = f2bfu(t.z) | ((unsigned int)f2bfu(t.w) << 16);
            *reinterpret_cast<uint2*>(&Bs[brow * 32 + c]) = make_uint2(lo, hi);
        }
        __syncthreads();

        v16bf af[2], bfb[2];
        af[0]  = load_frag_a(As, wm * 32 + 0,  32, 0, lane);
        af[1]  = load_frag_a(As, wm * 32 + 16, 32, 0, lane);
        bfb[0] = load_frag_b_nk(Bs, wn * 32 + 0,  32, 0, lane);
        bfb[1] = load_frag_b_nk(Bs, wn * 32 + 16, 32, 0, lane);
#pragma unroll
        for (int mt = 0; mt < 2; ++mt)
#pragma unroll
            for (int nt = 0; nt < 2; ++nt)
                acc[mt][nt] = wmma_bf16(af[mt], bfb[nt], acc[mt][nt]);
        __syncthreads();
    }

    // epilogue: + bias, bf16, scatter to [B,H,S,HD]
#pragma unroll
    for (int mt = 0; mt < 2; ++mt) {
#pragma unroll
        for (int nt = 0; nt < 2; ++nt) {
            int gm0 = m0 + wm * 32 + mt * 16 + (lane >> 4) * 8;
            int gn  = n0 + wn * 32 + nt * 16 + (lane & 15);
            float bv_ = bias[gn];
            int hh = gn >> 6, hd = gn & 63;
#pragma unroll
            for (int r = 0; r < 8; ++r) {
                int row = gm0 + r;
                int bb = row >> 11, ss = row & 2047;
                outp[((size_t)(bb * Hc + hh) * Sc + ss) * HDc + hd] =
                    f2bfu(acc[mt][nt][r] + bv_);
            }
        }
    }
}

// ===========================================================================
// Kernel 2: column-softmax stats.  For each (b,h) and k-column block (64),
// iterate all q-tiles: s = (Q K^T)/8 via WMMA, then online m[k], Z[k].
// grid = (S/64=32 kblocks, B*H=32), block = 256.
// ===========================================================================
__global__ __launch_bounds__(256) void k_stats(
    const unsigned short* __restrict__ qb, const unsigned short* __restrict__ kb,
    float* __restrict__ mOut, float* __restrict__ rzOut)
{
    __shared__ unsigned short Qs[64 * 64];
    __shared__ unsigned short Ks[64 * 64];
    __shared__ float Scr[64][65];
    __shared__ float Mcol[64];
    __shared__ float Zcol[64];

    const int tid  = threadIdx.x;
    const int lane = tid & 31;
    const int wave = tid >> 5;
    const int wm   = wave >> 1;     // q rows: wm*16
    const int wn   = wave & 1;      // k cols: wn*32
    const int kblk = blockIdx.x;
    const int bh   = blockIdx.y;

    // stage K block once via async Global->LDS (8 KB, 2 x b128 per thread)
    {
        const unsigned short* src = kb + ((size_t)bh * Sc + kblk * 64) * HDc;
        async_ld_b128(&Ks[tid * 8],         src + tid * 8);
        async_ld_b128(&Ks[(tid + 256) * 8], src + (tid + 256) * 8);
    }
    if (tid < 64) { Mcol[tid] = -1e30f; Zcol[tid] = 0.0f; }

    for (int q0 = 0; q0 < Sc; q0 += 64) {
        const unsigned short* qsrc = qb + ((size_t)bh * Sc + q0) * HDc;
        async_ld_b128(&Qs[tid * 8],         qsrc + tid * 8);
        async_ld_b128(&Qs[(tid + 256) * 8], qsrc + (tid + 256) * 8);
        wait_async0();
        __syncthreads();

        v8f sacc[2] = { vzero8(), vzero8() };
#pragma unroll
        for (int kk = 0; kk < 64; kk += 32) {
            v16bf af = load_frag_a(Qs, wm * 16, 64, kk, lane);
#pragma unroll
            for (int nt = 0; nt < 2; ++nt) {
                v16bf bfb = load_frag_b_nk(Ks, wn * 32 + nt * 16, 64, kk, lane);
                sacc[nt] = wmma_bf16(af, bfb, sacc[nt]);
            }
        }
#pragma unroll
        for (int nt = 0; nt < 2; ++nt) {
            int col  = wn * 32 + nt * 16 + (lane & 15);
            int row0 = wm * 16 + (lane >> 4) * 8;
#pragma unroll
            for (int r = 0; r < 8; ++r) Scr[row0 + r][col] = sacc[nt][r] * 0.125f;
        }
        __syncthreads();

        if (tid < 64) {     // one thread owns one k column; online max/sum
            int col = tid;
            float tmax = -1e30f;
            for (int r = 0; r < 64; ++r) tmax = fmaxf(tmax, Scr[r][col]);
            float mOld = Mcol[col];
            float mNew = fmaxf(mOld, tmax);
            float z = Zcol[col] * __expf(mOld - mNew);
            for (int r = 0; r < 64; ++r) z += __expf(Scr[r][col] - mNew);
            Mcol[col] = mNew; Zcol[col] = z;
        }
        __syncthreads();
    }

    if (tid < 64) {
        size_t o = (size_t)bh * Sc + kblk * 64 + tid;
        mOut[o]  = Mcol[tid];
        rzOut[o] = 1.0f / Zcol[tid];
    }
}

// ===========================================================================
// Kernel 3: apply.  For each (b,h) and q-block (64): loop k-blocks, recompute
// scores via WMMA, p = exp(s-m[k])/Z[k] (bf16 via LDS), weighted += p @ V via
// WMMA.  Writes bf16 ctx in [B,S,D] (col = h*64+hd).
// grid = (S/64=32 qblocks, B*H=32), block = 256.
// ===========================================================================
__global__ __launch_bounds__(256) void k_apply(
    const unsigned short* __restrict__ qb, const unsigned short* __restrict__ kb,
    const unsigned short* __restrict__ vb,
    const float* __restrict__ mSt, const float* __restrict__ rzSt,
    unsigned short* __restrict__ ctx)
{
    __shared__ unsigned short Qs[64 * 64];
    __shared__ unsigned short Ks[64 * 64];
    __shared__ unsigned short Vs[64 * 64];
    __shared__ unsigned short Ps[64 * 64];
    __shared__ float Mcol[64];
    __shared__ float Rz[64];

    const int tid  = threadIdx.x;
    const int lane = tid & 31;
    const int wave = tid >> 5;
    const int wm   = wave >> 1;     // q rows: wm*16
    const int wn   = wave & 1;      // cols (kcol for scores / hd for output): wn*32
    const int qblk = blockIdx.x;
    const int bh   = blockIdx.y;

    // stage Q block once (async; first in-loop wait covers it)
    {
        const unsigned short* src = qb + ((size_t)bh * Sc + qblk * 64) * HDc;
        async_ld_b128(&Qs[tid * 8],         src + tid * 8);
        async_ld_b128(&Qs[(tid + 256) * 8], src + (tid + 256) * 8);
    }

    v8f wacc[2] = { vzero8(), vzero8() };

    for (int kb0 = 0; kb0 < Sc; kb0 += 64) {
        __syncthreads();   // protect Ks/Vs/Ps readers of previous iteration
        {
            const unsigned short* ks = kb + ((size_t)bh * Sc + kb0) * HDc;
            const unsigned short* vs = vb + ((size_t)bh * Sc + kb0) * HDc;
            async_ld_b128(&Ks[tid * 8],         ks + tid * 8);
            async_ld_b128(&Ks[(tid + 256) * 8], ks + (tid + 256) * 8);
            async_ld_b128(&Vs[tid * 8],         vs + tid * 8);
            async_ld_b128(&Vs[(tid + 256) * 8], vs + (tid + 256) * 8);
        }
        if (tid < 64) {
            size_t o = (size_t)bh * Sc + kb0 + tid;
            Mcol[tid] = mSt[o];
            Rz[tid]   = rzSt[o];
        }
        wait_async0();
        __syncthreads();

        // scores tile (64q x 64k)
        v8f sacc[2] = { vzero8(), vzero8() };
#pragma unroll
        for (int kk = 0; kk < 64; kk += 32) {
            v16bf af = load_frag_a(Qs, wm * 16, 64, kk, lane);
#pragma unroll
            for (int nt = 0; nt < 2; ++nt) {
                v16bf bfb = load_frag_b_nk(Ks, wn * 32 + nt * 16, 64, kk, lane);
                sacc[nt] = wmma_bf16(af, bfb, sacc[nt]);
            }
        }
        // probs -> bf16 in LDS
#pragma unroll
        for (int nt = 0; nt < 2; ++nt) {
            int col  = wn * 32 + nt * 16 + (lane & 15);
            float mv = Mcol[col], rz = Rz[col];
            int row0 = wm * 16 + (lane >> 4) * 8;
#pragma unroll
            for (int r = 0; r < 8; ++r) {
                float p = __expf(sacc[nt][r] * 0.125f - mv) * rz;
                Ps[(row0 + r) * 64 + col] = f2bfu(p);
            }
        }
        __syncthreads();

        // weighted += P(64x64) @ V(64x64)
#pragma unroll
        for (int kk = 0; kk < 64; kk += 32) {
            v16bf af = load_frag_a(Ps, wm * 16, 64, kk, lane);
#pragma unroll
            for (int nt = 0; nt < 2; ++nt) {
                v16bf bfb = load_frag_b_kn(Vs, kk, 64, wn * 32 + nt * 16, lane);
                wacc[nt] = wmma_bf16(af, bfb, wacc[nt]);
            }
        }
    }

    // epilogue: ctx[b*S+s][h*64+hd] (bf16)
    const int b_ = bh >> 4, h_ = bh & 15;
#pragma unroll
    for (int nt = 0; nt < 2; ++nt) {
        int hd  = wn * 32 + nt * 16 + (lane & 15);
        int col = h_ * HDc + hd;
        int s0  = qblk * 64 + wm * 16 + (lane >> 4) * 8;
#pragma unroll
        for (int r = 0; r < 8; ++r)
            ctx[((size_t)(b_ * Sc + s0 + r)) * Dc + col] = f2bfu(wacc[nt][r]);
    }
}

// ===========================================================================
// Kernel 4: output projection.  out = ctx @ wo^T + bo (fp32 out).
// grid = (D/64=16, N/128=32), block = 256.
// ===========================================================================
__global__ __launch_bounds__(256) void k_oproj(
    const unsigned short* __restrict__ ctx, const float* __restrict__ wo,
    const float* __restrict__ bo, float* __restrict__ out)
{
    __shared__ unsigned short As[128 * 32];
    __shared__ unsigned short Bs[64 * 32];

    const int tid  = threadIdx.x;
    const int lane = tid & 31;
    const int wave = tid >> 5;
    const int wm   = wave >> 1;
    const int wn   = wave & 1;
    const int n0   = blockIdx.x * 64;
    const int m0   = blockIdx.y * 128;

    v8f acc[2][2];
#pragma unroll
    for (int mt = 0; mt < 2; ++mt)
#pragma unroll
        for (int nt = 0; nt < 2; ++nt) acc[mt][nt] = vzero8();

    const int arow = tid >> 1, ah = tid & 1;
    const int brow = tid >> 2, bq4 = tid & 3;

    for (int k0 = 0; k0 < Dc; k0 += 32) {
        // stage A from bf16 ctx via async Global->LDS (straight byte copy)
#pragma unroll
        for (int i = 0; i < 2; ++i) {
            int c = ah * 16 + i * 8;
            async_ld_b128(&As[arow * 32 + c],
                          ctx + (size_t)(m0 + arow) * Dc + k0 + c);
        }
        if (k0 + 32 < Dc)
            __builtin_prefetch(&wo[(size_t)(n0 + brow) * Dc + k0 + 32 + bq4 * 8], 0, 3);
        // stage Bt[n][k] = wo[n0+n][k0+k], fp32 -> bf16
#pragma unroll
        for (int i = 0; i < 2; ++i) {
            int c = bq4 * 8 + i * 4;
            float4 t = *reinterpret_cast<const float4*>(&wo[(size_t)(n0 + brow) * Dc + k0 + c]);
            unsigned int lo = f2bfu(t.x) | ((unsigned int)f2bfu(t.y) << 16);
            unsigned int hi = f2bfu(t.z) | ((unsigned int)f2bfu(t.w) << 16);
            *reinterpret_cast<uint2*>(&Bs[brow * 32 + c]) = make_uint2(lo, hi);
        }
        wait_async0();
        __syncthreads();

        v16bf af[2], bfb[2];
        af[0]  = load_frag_a(As, wm * 32 + 0,  32, 0, lane);
        af[1]  = load_frag_a(As, wm * 32 + 16, 32, 0, lane);
        bfb[0] = load_frag_b_nk(Bs, wn * 32 + 0,  32, 0, lane);
        bfb[1] = load_frag_b_nk(Bs, wn * 32 + 16, 32, 0, lane);
#pragma unroll
        for (int mt = 0; mt < 2; ++mt)
#pragma unroll
            for (int nt = 0; nt < 2; ++nt)
                acc[mt][nt] = wmma_bf16(af[mt], bfb[nt], acc[mt][nt]);
        __syncthreads();
    }

#pragma unroll
    for (int mt = 0; mt < 2; ++mt) {
#pragma unroll
        for (int nt = 0; nt < 2; ++nt) {
            int gm0 = m0 + wm * 32 + mt * 16 + (lane >> 4) * 8;
            int gn  = n0 + wn * 32 + nt * 16 + (lane & 15);
            float bv_ = bo[gn];
#pragma unroll
            for (int r = 0; r < 8; ++r)
                out[(size_t)(gm0 + r) * Dc + gn] = acc[mt][nt][r] + bv_;
        }
    }
}

// ===========================================================================
// Host launcher.  Workspace layout:
//   qb/kb/vb [B*H*S*HD] bf16 : 8 MiB each
//   ctx      [B*S*D]    bf16 : 8 MiB
//   m, rz    [B*H*S]    f32  : 256 KiB each      total ~32.5 MiB
// ===========================================================================
extern "C" void kernel_launch(void* const* d_in, const int* in_sizes, int n_in,
                              void* d_out, int out_size, void* d_ws, size_t ws_size,
                              hipStream_t stream) {
    const float* x  = (const float*)d_in[0];
    const float* wq = (const float*)d_in[1];
    const float* bq = (const float*)d_in[2];
    const float* wk = (const float*)d_in[3];
    const float* bk = (const float*)d_in[4];
    const float* wv = (const float*)d_in[5];
    const float* bv = (const float*)d_in[6];
    const float* wo = (const float*)d_in[7];
    const float* bo = (const float*)d_in[8];
    float* out = (float*)d_out;

    const size_t elems = (size_t)Bc * Hc * Sc * HDc;   // 4 Mi elements
    unsigned short* qb  = (unsigned short*)d_ws;
    unsigned short* kb  = qb + elems;
    unsigned short* vb  = kb + elems;
    unsigned short* ctx = vb + elems;
    float* mSt  = (float*)(ctx + elems);
    float* rzSt = mSt + (size_t)Bc * Hc * Sc;

    k_qkv  <<<dim3(16, 32, 3), 256, 0, stream>>>(x, wq, bq, wk, bk, wv, bv, qb, kb, vb);
    k_stats<<<dim3(32, 32),    256, 0, stream>>>(qb, kb, mSt, rzSt);
    k_apply<<<dim3(32, 32),    256, 0, stream>>>(qb, kb, vb, mSt, rzSt, ctx);
    k_oproj<<<dim3(16, 32),    256, 0, stream>>>(ctx, wo, bo, out);

    (void)in_sizes; (void)n_in; (void)out_size; (void)ws_size;
}
